// TransformerBlock_34093450395992
// MI455X (gfx1250) — matrix operand order repaired and checked by reading, not compiled, not run on hardware
//
#include <hip/hip_runtime.h>

// ---------------------------------------------------------------------------
// Types for CDNA5 WMMA (wave32, 16x16x32 bf16 -> f32)
// ---------------------------------------------------------------------------
typedef __bf16 bf16;
typedef __attribute__((ext_vector_type(16))) __bf16 v16bf;
typedef __attribute__((ext_vector_type(8)))  float  v8f;

union Frag {
    v16bf v;
    uint4 q[2];   // 32 bytes = 16 bf16
};

__device__ __forceinline__ v8f wmma_bf16(const Frag& a, const Frag& b, v8f c) {
    return __builtin_amdgcn_wmma_f32_16x16x32_bf16(
        /*neg_a=*/false, a.v, /*neg_b=*/false, b.v,
        /*c_mod=*/(short)0, c, /*reuse_a=*/false, /*reuse_b=*/false);
}

#define SEQ 4096
#define NHEAD 12

// ---------------------------------------------------------------------------
// Weight conversion: fp32 [K,N] row-major -> bf16 packed [K/32][N][32]
// so that a B-matrix WMMA fragment (lane holds 16 contiguous K values for one
// column) is two contiguous 16-byte loads.
// ---------------------------------------------------------------------------
__global__ void cvt_weight(const float* __restrict__ W, bf16* __restrict__ Wp,
                           int K, int N) {
    int idx = blockIdx.x * 256 + threadIdx.x;
    if (idx >= K * N) return;
    int k = idx / N;
    int n = idx - k * N;
    Wp[(size_t)(k >> 5) * ((size_t)N * 32) + (size_t)n * 32 + (k & 31)] =
        (bf16)W[idx];
}

// ---------------------------------------------------------------------------
// LayerNorm: fp32 in, bf16 out (row-major), one wave (32 lanes) per row.
// ---------------------------------------------------------------------------
__global__ __launch_bounds__(256)
void layernorm_bf16(const float* __restrict__ x, const float* __restrict__ sc,
                    const float* __restrict__ sh, bf16* __restrict__ out, int D) {
    int row  = blockIdx.x * 8 + (threadIdx.x >> 5);
    int lane = threadIdx.x & 31;
    const float* xr = x + (size_t)row * D;
    float s = 0.f, s2 = 0.f;
    for (int i = lane; i < D; i += 32) { float v = xr[i]; s += v; s2 += v * v; }
    for (int o = 16; o > 0; o >>= 1) {
        s  += __shfl_xor(s,  o, 32);
        s2 += __shfl_xor(s2, o, 32);
    }
    float mean = s / (float)D;
    float var  = s2 / (float)D - mean * mean;
    float inv  = rsqrtf(var + 1e-5f);
    bf16* orow = out + (size_t)row * D;
    for (int i = lane; i < D; i += 32)
        orow[i] = (bf16)(sc[i] * ((xr[i] - mean) * inv) + sh[i]);
}

// ---------------------------------------------------------------------------
// bf16 GEMM with fused epilogue.
//   C[M,N] = A[M,K](bf16 row-major) @ Bp (bf16 packed [K/32][N][32])
//   + bias (opt) -> gelu (opt) -> + residual fp32 (opt)
//   writes fp32 (opt) and/or bf16 (opt).
// vpack != 0: bf16 output scattered into attention-V packed layout
//   [b][h][s/32][d(64)][key(32)]  (N must be 768, M tokens of 2x4096).
// Block: 256 threads = 8 waves; tile 128(M) x 128(N); wave tile 64x32;
// per-wave 4x2 accumulator tiles of v_wmma_f32_16x16x32_bf16.
// Requires: M%128==0, N%128==0, K%32==0 (true for all uses here).
// ---------------------------------------------------------------------------
__global__ __launch_bounds__(256)
void gemm_bf16(const bf16* __restrict__ A, const bf16* __restrict__ Bp,
               const float* __restrict__ bias, const float* __restrict__ res,
               float* __restrict__ outF, bf16* __restrict__ outB,
               int M, int N, int K, int dogelu, int vpack) {
    const int lane = threadIdx.x & 31;
    const int w    = threadIdx.x >> 5;
    const int wm   = w & 1;        // 2 waves along M
    const int wn   = w >> 1;       // 4 waves along N
    const int lm   = lane & 15;
    const int hi   = lane >> 4;    // 0: lanes 0-15, 1: lanes 16-31
    const int bm   = blockIdx.x * 128;
    const int bn   = blockIdx.y * 128;

    v8f acc[4][2];
    #pragma unroll
    for (int i = 0; i < 4; ++i)
        #pragma unroll
        for (int j = 0; j < 2; ++j) acc[i][j] = (v8f){0,0,0,0,0,0,0,0};

    // A fragment base element offsets (row-major): row*K + (hi?16:0)
    size_t aoff[4];
    #pragma unroll
    for (int i = 0; i < 4; ++i) {
        int row = bm + wm * 64 + i * 16 + lm;
        aoff[i] = (size_t)row * K + (size_t)(hi * 16);
    }
    // B fragment base: packed [kb][col][32] ; col fixed per lane
    size_t boff[2];
    #pragma unroll
    for (int j = 0; j < 2; ++j) {
        int col = bn + wn * 32 + j * 16 + lm;
        boff[j] = (size_t)col * 32 + (size_t)(hi * 16);
    }
    const size_t bstride = (size_t)N * 32;   // elements per K-block of 32

    for (int k0 = 0; k0 < K; k0 += 32) {
        Frag a[4], b[2];
        #pragma unroll
        for (int i = 0; i < 4; ++i) {
            const uint4* p = (const uint4*)(A + aoff[i] + k0);
            a[i].q[0] = p[0];   // K = k0 + hi*16 + [0..7]
            a[i].q[1] = p[1];   // K = k0 + hi*16 + [8..15]
        }
        const size_t bb = (size_t)(k0 >> 5) * bstride;
        #pragma unroll
        for (int j = 0; j < 2; ++j) {
            const uint4* p = (const uint4*)(Bp + bb + boff[j]);
            b[j].q[0] = p[0];   // K = k0 + hi*16 + [0..7]
            b[j].q[1] = p[1];   // K = k0 + hi*16 + [8..15]
        }
        #pragma unroll
        for (int i = 0; i < 4; ++i)
            #pragma unroll
            for (int j = 0; j < 2; ++j)
                acc[i][j] = wmma_bf16(a[i], b[j], acc[i][j]);
    }

    // Epilogue. C layout: VGPR g -> row_local = g + hi*8, col across lanes.
    #pragma unroll
    for (int i = 0; i < 4; ++i) {
        #pragma unroll
        for (int j = 0; j < 2; ++j) {
            int col = bn + wn * 32 + j * 16 + lm;
            #pragma unroll
            for (int g = 0; g < 8; ++g) {
                int row = bm + wm * 64 + i * 16 + g + hi * 8;
                float v = acc[i][j][g];
                if (bias) v += bias[col];
                if (dogelu) {
                    float t = v + 0.044715f * v * v * v;
                    v = 0.5f * v * (1.0f + tanhf(0.7978845608028654f * t));
                }
                size_t o = (size_t)row * N + col;
                if (res)  v += res[o];
                if (outF) outF[o] = v;
                if (outB) {
                    if (vpack) {
                        // token -> (b,s); col -> (h,d); pack [b][h][s/32][d][key]
                        int bb2 = row >> 12;          // row / 4096
                        int s   = row & 4095;
                        int hh  = col >> 6;
                        int d   = col & 63;
                        size_t po = ((size_t)((bb2 * NHEAD + hh) * (SEQ / 32) +
                                              (s >> 5)) * 2048) +
                                    (size_t)d * 32 + (s & 31);
                        outB[po] = (bf16)v;
                    } else {
                        outB[o] = (bf16)v;
                    }
                }
            }
        }
    }
}

// ---------------------------------------------------------------------------
// Flash attention (causal), head_dim=64, S=4096, 12 heads.
// Q,K: bf16 [B*S, 768] token-major (head h occupies cols h*64..h*64+63).
// Vp: bf16 packed [b][h][s/32][d(64)][key(32)] -> PV B-fragments are two
//     contiguous 16B global loads (L2-resident).
// O: bf16 [B*S, 768].
// Grid: (S/128, B*HEADS). Block 256 threads = 8 fully independent waves,
// each wave owns 16 query rows. No block barriers (waves diverge in trip
// count); same-wave LDS ordering is guaranteed by DScnt in-order semantics.
// ---------------------------------------------------------------------------
__global__ __launch_bounds__(256)
void attn_kernel(const bf16* __restrict__ Q, const bf16* __restrict__ Kb,
                 const bf16* __restrict__ Vp, bf16* __restrict__ O) {
    __shared__ bf16 Ps[8][16 * 32];   // per-wave P tile, row-major [q][key]

    const int lane = threadIdx.x & 31;
    const int w    = threadIdx.x >> 5;
    const int lm   = lane & 15;
    const int hi   = lane >> 4;
    const int bh   = blockIdx.y;
    const int b    = bh / NHEAD;
    const int h    = bh - b * NHEAD;
    const int q0   = blockIdx.x * 128 + w * 16;

    const size_t base  = (size_t)b * SEQ * 768 + (size_t)h * 64;
    const bf16*  vbase = Vp + (size_t)bh * (SEQ / 32) * 2048 +
                         (size_t)lm * 32 + (size_t)hi * 16;   // d=lm (+n*16)

    // Q as two A-fragments (d-chunks 0..31 and 32..63), persistent.
    Frag qa[2];
    {
        const bf16* qp = Q + base + (size_t)(q0 + lm) * 768;
        #pragma unroll
        for (int c = 0; c < 2; ++c) {
            const uint4* p = (const uint4*)(qp + c * 32 + hi * 16);
            qa[c].q[0] = p[0];
            qa[c].q[1] = p[1];
        }
    }

    v8f acc[4];
    #pragma unroll
    for (int n = 0; n < 4; ++n) acc[n] = (v8f){0,0,0,0,0,0,0,0};
    float mrow[8], lrow[8];
    #pragma unroll
    for (int g = 0; g < 8; ++g) { mrow[g] = -1e30f; lrow[g] = 0.f; }

    const int nkt = q0 / 32 + 1;   // causal: key tiles covering 0..q0+15
    for (int kt = 0; kt < nkt; ++kt) {
        const int kb0 = kt * 32;

        // Scores: S(16x32) = Q(16x64) @ K^T(64x32): two 16-col tiles.
        v8f s[2];
        #pragma unroll
        for (int j = 0; j < 2; ++j) {
            const bf16* kp = Kb + base + (size_t)(kb0 + j * 16 + lm) * 768;
            Frag kf[2];
            #pragma unroll
            for (int c = 0; c < 2; ++c) {
                const uint4* p = (const uint4*)(kp + c * 32 + hi * 16);
                kf[c].q[0] = p[0];
                kf[c].q[1] = p[1];
            }
            v8f z = (v8f){0,0,0,0,0,0,0,0};
            z = wmma_bf16(qa[0], kf[0], z);
            s[j] = wmma_bf16(qa[1], kf[1], z);
        }

        // Online softmax update (fp32).
        float pr[2][8];
        #pragma unroll
        for (int g = 0; g < 8; ++g) {
            const int qrow = q0 + g + hi * 8;
            float a0 = s[0][g] * 0.125f;      // 1/sqrt(64)
            float a1 = s[1][g] * 0.125f;
            if (kb0 + lm      > qrow) a0 = -1e30f;
            if (kb0 + 16 + lm > qrow) a1 = -1e30f;
            pr[0][g] = a0; pr[1][g] = a1;
            float mx = fmaxf(a0, a1);
            #pragma unroll
            for (int o = 8; o > 0; o >>= 1) mx = fmaxf(mx, __shfl_xor(mx, o, 32));
            float mnew  = fmaxf(mrow[g], mx);
            float alpha = __expf(mrow[g] - mnew);
            mrow[g] = mnew;
            float p0 = __expf(pr[0][g] - mnew);
            float p1 = __expf(pr[1][g] - mnew);
            pr[0][g] = p0; pr[1][g] = p1;
            float rs = p0 + p1;
            #pragma unroll
            for (int o = 8; o > 0; o >>= 1) rs += __shfl_xor(rs, o, 32);
            lrow[g] = lrow[g] * alpha + rs;
            #pragma unroll
            for (int n = 0; n < 4; ++n) acc[n][g] *= alpha;
        }

        // P (C-layout) -> LDS -> A-fragment layout.
        #pragma unroll
        for (int g = 0; g < 8; ++g) {
            const int rl = g + hi * 8;
            Ps[w][rl * 32 + lm]      = (bf16)pr[0][g];
            Ps[w][rl * 32 + 16 + lm] = (bf16)pr[1][g];
        }
        Frag pa;
        {
            const uint4* p = (const uint4*)(&Ps[w][lm * 32 + hi * 16]);
            pa.q[0] = p[0];
            pa.q[1] = p[1];
        }

        // ctx += P(16x32) @ V(32x64): 4 WMMA, B-fragments direct from
        // packed global V (two contiguous b128 loads each).
        const bf16* vt = vbase + (size_t)kt * 2048;
        #pragma unroll
        for (int n = 0; n < 4; ++n) {
            Frag vb;
            const uint4* p = (const uint4*)(vt + (size_t)n * 16 * 32);
            vb.q[0] = p[0];   // key = hi*16 + [0..7]
            vb.q[1] = p[1];   // key = hi*16 + [8..15]
            acc[n] = wmma_bf16(pa, vb, acc[n]);
        }
    }

    // Finalize: divide by l, store bf16 ctx.
    #pragma unroll
    for (int n = 0; n < 4; ++n) {
        #pragma unroll
        for (int g = 0; g < 8; ++g) {
            const int qrow = q0 + g + hi * 8;
            O[base + (size_t)qrow * 768 + n * 16 + lm] =
                (bf16)(acc[n][g] / lrow[g]);
        }
    }
}

// ---------------------------------------------------------------------------
// Host: orchestrate the transformer block.
// ---------------------------------------------------------------------------
extern "C" void kernel_launch(void* const* d_in, const int* in_sizes, int n_in,
                              void* d_out, int out_size, void* d_ws, size_t ws_size,
                              hipStream_t stream) {
    const float* x    = (const float*)d_in[0];
    const float* Wq   = (const float*)d_in[1];
    const float* Wk   = (const float*)d_in[2];
    const float* Wv   = (const float*)d_in[3];
    const float* Wo   = (const float*)d_in[4];
    const float* bo   = (const float*)d_in[5];
    const float* W1   = (const float*)d_in[6];
    const float* b1   = (const float*)d_in[7];
    const float* W2   = (const float*)d_in[8];
    const float* b2   = (const float*)d_in[9];
    const float* ln1s = (const float*)d_in[10];
    const float* ln1b = (const float*)d_in[11];
    const float* ln2s = (const float*)d_in[12];
    const float* ln2b = (const float*)d_in[13];

    const int D = 768, H4 = 3072, NT = 8192;  // tokens = B*S

    char* ws = (char*)d_ws;
    size_t off = 0;
    auto take = [&](size_t bytes) {
        size_t r = off;
        off += (bytes + 255) & ~(size_t)255;
        return r;
    };
    bf16* wqp = (bf16*)(ws + take((size_t)D * D * 2));
    bf16* wkp = (bf16*)(ws + take((size_t)D * D * 2));
    bf16* wvp = (bf16*)(ws + take((size_t)D * D * 2));
    bf16* wop = (bf16*)(ws + take((size_t)D * D * 2));
    bf16* w1p = (bf16*)(ws + take((size_t)D * H4 * 2));
    bf16* w2p = (bf16*)(ws + take((size_t)H4 * D * 2));
    bf16* hbuf = (bf16*)(ws + take((size_t)NT * D * 2));
    size_t qoff = take((size_t)NT * D * 2);
    bf16* qbuf = (bf16*)(ws + qoff);
    bf16* kbuf = (bf16*)(ws + take((size_t)NT * D * 2));
    bf16* vbuf = (bf16*)(ws + take((size_t)NT * D * 2));  // packed V layout
    bf16* ctxb = (bf16*)(ws + take((size_t)NT * D * 2));
    float* x1  = (float*)(ws + take((size_t)NT * D * 4));
    bf16* gbuf = (bf16*)(ws + qoff);   // reuse q/k/v/ctx region (dead by then)

    // 1) weights -> packed bf16
    cvt_weight<<<(D * D + 255) / 256, 256, 0, stream>>>(Wq, wqp, D, D);
    cvt_weight<<<(D * D + 255) / 256, 256, 0, stream>>>(Wk, wkp, D, D);
    cvt_weight<<<(D * D + 255) / 256, 256, 0, stream>>>(Wv, wvp, D, D);
    cvt_weight<<<(D * D + 255) / 256, 256, 0, stream>>>(Wo, wop, D, D);
    cvt_weight<<<(D * H4 + 255) / 256, 256, 0, stream>>>(W1, w1p, D, H4);
    cvt_weight<<<(H4 * D + 255) / 256, 256, 0, stream>>>(W2, w2p, H4, D);

    // 2) LN1
    layernorm_bf16<<<NT / 8, 256, 0, stream>>>(x, ln1s, ln1b, hbuf, D);

    // 3) q,k projections row-major bf16; v projection in packed layout
    gemm_bf16<<<dim3(64, 6), 256, 0, stream>>>(hbuf, wqp, nullptr, nullptr,
                                               nullptr, qbuf, NT, D, D, 0, 0);
    gemm_bf16<<<dim3(64, 6), 256, 0, stream>>>(hbuf, wkp, nullptr, nullptr,
                                               nullptr, kbuf, NT, D, D, 0, 0);
    gemm_bf16<<<dim3(64, 6), 256, 0, stream>>>(hbuf, wvp, nullptr, nullptr,
                                               nullptr, vbuf, NT, D, D, 0, 1);

    // 4) causal flash attention
    attn_kernel<<<dim3(SEQ / 128, 2 * NHEAD), 256, 0, stream>>>(qbuf, kbuf,
                                                                vbuf, ctxb);

    // 5) out-proj + bias + residual -> x1 (fp32)
    gemm_bf16<<<dim3(64, 6), 256, 0, stream>>>(ctxb, wop, bo, x,
                                               x1, nullptr, NT, D, D, 0, 0);

    // 6) LN2
    layernorm_bf16<<<NT / 8, 256, 0, stream>>>(x1, ln2s, ln2b, hbuf, D);

    // 7) MLP up + bias + GELU (bf16 out)
    gemm_bf16<<<dim3(64, 24), 256, 0, stream>>>(hbuf, w1p, b1, nullptr,
                                                nullptr, gbuf, NT, H4, D, 1, 0);

    // 8) MLP down + bias + residual -> final output (fp32)
    gemm_bf16<<<dim3(64, 6), 256, 0, stream>>>(gbuf, w2p, b2, x1,
                                               (float*)d_out, nullptr,
                                               NT, D, H4, 0, 0);
}
